// MentionPruner_16131897163798
// MI455X (gfx1250) — matrix operand order repaired and checked by reading, not compiled, not run on hardware
//
#include <hip/hip_runtime.h>
#include <hip/hip_bf16.h>

#define BB 8
#define SS 512
#define LLEN 15
#define DD 768
#define HH 512
#define KKP 102
#define NSPANS (SS*LLEN)      // 7680
#define MTOT (BB*NSPANS)      // 61440
#define MT   64               // M-tile rows per workgroup
#define KCH  128              // K staging chunk width (bf16)

typedef __attribute__((ext_vector_type(16))) __bf16 v16bf;
typedef __attribute__((ext_vector_type(8)))  __bf16 v8bf;
typedef __attribute__((ext_vector_type(8)))  float  v8f;

union AFrag { v16bf v; v8bf h[2]; };

// ---------------------------------------------------------------------------
// Prep: fp32 weights -> bf16 copies in workspace (L2-resident thereafter)
// ---------------------------------------------------------------------------
__global__ __launch_bounds__(256)
void mp_prep_weights(const float* __restrict__ w1, const float* __restrict__ w2,
                     __hip_bfloat16* __restrict__ w1b, __hip_bfloat16* __restrict__ w2b) {
    int i = blockIdx.x * 256 + threadIdx.x;
    if (i < DD * HH) w1b[i] = __float2bfloat16(w1[i]);
    if (i < HH * HH) w2b[i] = __float2bfloat16(w2[i]);
}

// ---------------------------------------------------------------------------
// Fused 3-layer MLP scorer. One block = 64 M-rows; 8 waves, each wave owns a
// 64-column slice and ALL four 16-row M-subtiles (16 WMMA accumulators), so
// every B fragment fetched from L2 feeds 4 WMMAs.
// ---------------------------------------------------------------------------
__global__ __launch_bounds__(256)
void mp_scorer(const float* __restrict__ A,          // [MTOT, DD] span_vecs
               const float* __restrict__ mask,       // [MTOT] span_mask
               const __hip_bfloat16* __restrict__ w1bh,
               const float* __restrict__ b1,
               const __hip_bfloat16* __restrict__ w2bh,
               const float* __restrict__ b2,
               const float* __restrict__ w3,
               const float* __restrict__ b3,
               float* __restrict__ out_scores)       // [MTOT]
{
    const __bf16* __restrict__ w1b = reinterpret_cast<const __bf16*>(w1bh);
    const __bf16* __restrict__ w2b = reinterpret_cast<const __bf16*>(w2bh);

    __shared__ __bf16 Abuf[MT][KCH];  // 16 KB: A K-chunk (bf16)
    __shared__ __bf16 Hbuf[MT][HH];   // 64 KB: h1 tile (bf16, post-ReLU)
    __shared__ float  ssc[MT];        // per-row layer-3 partial sums

    const int tid  = threadIdx.x;
    const int lane = tid & 31;
    const int wv   = tid >> 5;            // wave id 0..7 -> 64-col slice
    const int half = lane >> 4;           // A-frag K-half select
    const int mrow = lane & 15;           // A-frag row within subtile
    const int chi  = lane >> 4;           // C/D: lanes 16-31 -> M+8
    const int cn   = lane & 15;           // C/D: N within tile

    const size_t mBase = (size_t)blockIdx.x * MT;

    if (tid < MT) ssc[tid] = 0.0f;

    const v8f vzero = {0.f,0.f,0.f,0.f,0.f,0.f,0.f,0.f};

    // ---- GEMM1: h1 = relu(A @ w1 + b1), K = 768 = 6 chunks x 4 wmma-steps ----
    v8f acc[4][4];                        // [msub][ntile]
#pragma unroll
    for (int s = 0; s < 4; ++s)
#pragma unroll
        for (int t = 0; t < 4; ++t) acc[s][t] = vzero;

    for (int c = 0; c < DD; c += KCH) {
        __syncthreads();                  // protect Abuf reuse
        // stage fp32 -> bf16 chunk (coalesced: 128 consecutive cols per row)
        for (int e = tid; e < MT * KCH; e += 256) {
            int r = e >> 7, kk = e & (KCH - 1);
            Abuf[r][kk] = (__bf16)A[(mBase + r) * (size_t)DD + c + kk];
        }
        __syncthreads();
#pragma unroll
        for (int kc = 0; kc < KCH; kc += 32) {
            AFrag af[4];
#pragma unroll
            for (int s = 0; s < 4; ++s) {
                af[s].h[0] = *(const v8bf*)&Abuf[s * 16 + mrow][kc + half * 8];
                af[s].h[1] = *(const v8bf*)&Abuf[s * 16 + mrow][kc + 16 + half * 8];
            }
#pragma unroll
            for (int t = 0; t < 4; ++t) {
                const int n0 = wv * 64 + t * 16;
                v16bf bf;
                for (int i = 0; i < 16; ++i) bf[i] = w1b[(size_t)(c + kc + (i < 8 ? half * 8 + i : 16 + half * 8 + (i - 8))) * HH + n0 + mrow];
#pragma unroll
                for (int s = 0; s < 4; ++s)
                    acc[s][t] = __builtin_amdgcn_wmma_f32_16x16x32_bf16(
                        false, af[s].v, false, bf, (short)0, acc[s][t], false, false);
            }
        }
    }
    __syncthreads();
    // bias + ReLU -> Hbuf (bf16), per documented C/D layout
#pragma unroll
    for (int t = 0; t < 4; ++t) {
        const int n = wv * 64 + t * 16 + cn;
        const float bias = b1[n];
#pragma unroll
        for (int s = 0; s < 4; ++s) {
#pragma unroll
            for (int r = 0; r < 8; ++r) {
                float v = acc[s][t][r] + bias;
                v = v > 0.f ? v : 0.f;
                Hbuf[s * 16 + chi * 8 + r][n] = (__bf16)v;
            }
        }
    }
    __syncthreads();

    // ---- GEMM2: h2 = relu(h1 @ w2 + b2), K = 512 -> 16 wmma-steps ----
    v8f acc2[4][4];
#pragma unroll
    for (int s = 0; s < 4; ++s)
#pragma unroll
        for (int t = 0; t < 4; ++t) acc2[s][t] = vzero;

    for (int k0 = 0; k0 < HH; k0 += 32) {
        AFrag af[4];
#pragma unroll
        for (int s = 0; s < 4; ++s) {
            af[s].h[0] = *(const v8bf*)&Hbuf[s * 16 + mrow][k0 + half * 8];
            af[s].h[1] = *(const v8bf*)&Hbuf[s * 16 + mrow][k0 + 16 + half * 8];
        }
#pragma unroll
        for (int t = 0; t < 4; ++t) {
            const int n0 = wv * 64 + t * 16;
            v16bf bf;
            for (int i = 0; i < 16; ++i) bf[i] = w2b[(size_t)(k0 + (i < 8 ? half * 8 + i : 16 + half * 8 + (i - 8))) * HH + n0 + mrow];
#pragma unroll
            for (int s = 0; s < 4; ++s)
                acc2[s][t] = __builtin_amdgcn_wmma_f32_16x16x32_bf16(
                    false, af[s].v, false, bf, (short)0, acc2[s][t], false, false);
        }
    }

    // ---- Layer 3: score = h2 . w3 (h2 never leaves registers) ----
    float part[4][8];
#pragma unroll
    for (int s = 0; s < 4; ++s)
#pragma unroll
        for (int r = 0; r < 8; ++r) part[s][r] = 0.f;
#pragma unroll
    for (int t = 0; t < 4; ++t) {
        const int n = wv * 64 + t * 16 + cn;
        const float bias = b2[n];
        const float w3n  = w3[n];
#pragma unroll
        for (int s = 0; s < 4; ++s) {
#pragma unroll
            for (int r = 0; r < 8; ++r) {
                float v = acc2[s][t][r] + bias;
                v = v > 0.f ? v : 0.f;
                part[s][r] += v * w3n;
            }
        }
    }
#pragma unroll
    for (int s = 0; s < 4; ++s)
#pragma unroll
        for (int r = 0; r < 8; ++r)
            atomicAdd(&ssc[s * 16 + chi * 8 + r], part[s][r]);   // ds_add_f32
    __syncthreads();

    if (tid < MT) {
        const size_t row = mBase + tid;
        float sc = ssc[tid] + b3[0] - (1.0f - mask[row]) * 10000.0f;
        out_scores[row] = sc;
    }
}

// ---------------------------------------------------------------------------
// Top-K by rank counting (exact jax.lax.top_k tie-break: lower index wins).
// ---------------------------------------------------------------------------
__global__ __launch_bounds__(256)
void mp_topk_rank(const float* __restrict__ scores, int* __restrict__ flags) {
    __shared__ float s[NSPANS];   // 30 KB
    const int b     = blockIdx.x / 30;
    const int chunk = blockIdx.x % 30;
    const float* sb = scores + (size_t)b * NSPANS;
    for (int j = threadIdx.x; j < NSPANS; j += 256) s[j] = sb[j];
    __syncthreads();
    const int i = chunk * 256 + threadIdx.x;
    const float v = s[i];
    int cnt = 0;
    for (int j = 0; j < NSPANS; ++j) {
        const float x = s[j];
        cnt += (x > v) ? 1 : ((x == v && j < i) ? 1 : 0);
    }
    flags[(size_t)b * NSPANS + i] = (cnt < KKP) ? 1 : 0;
}

// ---------------------------------------------------------------------------
// Compact selected indices (ascending) + emit idx/f_scores/f_begin/f_end/len
// ---------------------------------------------------------------------------
__global__ __launch_bounds__(256)
void mp_topk_compact(const int* __restrict__ flags, const float* __restrict__ scores,
                     const int* __restrict__ seqlen,
                     int* __restrict__ idx_ws, int* __restrict__ sl_ws,
                     float* __restrict__ o_idx, float* __restrict__ o_fscore,
                     float* __restrict__ o_fbegin, float* __restrict__ o_fend,
                     float* __restrict__ o_slen) {
    __shared__ int cnts[256];
    __shared__ int offs[256];
    const int b = blockIdx.x;
    const int t = threadIdx.x;
    const int PER = NSPANS / 256;  // 30
    const int* fb = flags + (size_t)b * NSPANS;
    int c = 0;
    for (int e = 0; e < PER; ++e) c += fb[t * PER + e];
    cnts[t] = c;
    __syncthreads();
    if (t == 0) {
        int a = 0;
        for (int q = 0; q < 256; ++q) { offs[q] = a; a += cnts[q]; }
    }
    __syncthreads();
    int pos = offs[t];
    for (int e = 0; e < PER; ++e) {
        const int i = t * PER + e;
        if (fb[i]) {
            if (pos < KKP) {
                const int s = i / LLEN, l = i % LLEN;
                idx_ws[b * KKP + pos]   = i;
                o_idx[b * KKP + pos]    = (float)i;
                o_fscore[b * KKP + pos] = scores[(size_t)b * NSPANS + i];
                o_fbegin[b * KKP + pos] = (float)s;
                o_fend[b * KKP + pos]   = (float)(s + l);
            }
            ++pos;
        }
    }
    if (t == 0) {
        int sl = (int)(0.2f * (float)seqlen[b]);
        if (sl > KKP) sl = KKP;
        sl_ws[b] = sl;
        o_slen[b] = (float)sl;
    }
}

// ---------------------------------------------------------------------------
// Gather pruned span vectors: one block per (b, k) row
// ---------------------------------------------------------------------------
__global__ __launch_bounds__(256)
void mp_gather_vecs(const float* __restrict__ A, const int* __restrict__ idx_ws,
                    float* __restrict__ o_vecs) {
    const int blk = blockIdx.x;        // b*KKP + pos
    const int b   = blk / KKP;
    const int i   = idx_ws[blk];
    const float* src = A + ((size_t)b * NSPANS + i) * DD;
    float* dst = o_vecs + (size_t)blk * DD;
    for (int c = threadIdx.x; c < DD; c += 256) dst[c] = src[c];
}

// ---------------------------------------------------------------------------
// Square + triangular masks
// ---------------------------------------------------------------------------
__global__ __launch_bounds__(256)
void mp_masks(const int* __restrict__ sl_ws, float* __restrict__ o_sq,
              float* __restrict__ o_tri) {
    const int e = blockIdx.x * 256 + threadIdx.x;
    const int tot = BB * KKP * KKP;
    if (e >= tot) return;
    const int b = e / (KKP * KKP);
    const int rem = e % (KKP * KKP);
    const int r = rem / KKP, c = rem % KKP;
    const int sl = sl_ws[b];
    const float valid = (r < sl && c < sl) ? 1.f : 0.f;
    o_sq[e]  = valid;
    o_tri[e] = (c <= r) ? valid : 0.f;
}

// ---------------------------------------------------------------------------
extern "C" void kernel_launch(void* const* d_in, const int* in_sizes, int n_in,
                              void* d_out, int out_size, void* d_ws, size_t ws_size,
                              hipStream_t stream) {
    (void)in_sizes; (void)n_in; (void)out_size; (void)ws_size;

    const float* span_vecs = (const float*)d_in[0];   // [B,S,L,D]
    const float* span_mask = (const float*)d_in[1];   // [B,S,L]
    const int*   seqlen    = (const int*)d_in[4];     // [B]
    const float* w1 = (const float*)d_in[5];
    const float* b1 = (const float*)d_in[6];
    const float* w2 = (const float*)d_in[7];
    const float* b2 = (const float*)d_in[8];
    const float* w3 = (const float*)d_in[9];
    const float* b3 = (const float*)d_in[10];

    // Output tuple layout (flat f32, reference return order)
    float* out       = (float*)d_out;
    float* o_scores  = out;                          // [B,S,L,1]  61440
    float* o_idx     = o_scores + MTOT;              // [B,K]      816
    float* o_vecs    = o_idx + BB * KKP;             // [B,K,D]    626688
    float* o_fscore  = o_vecs + (size_t)BB * KKP * DD;
    float* o_fbegin  = o_fscore + BB * KKP;
    float* o_fend    = o_fbegin + BB * KKP;
    float* o_slen    = o_fend + BB * KKP;            // [B]
    float* o_sq      = o_slen + BB;                  // [B,K,K]
    float* o_tri     = o_sq + (size_t)BB * KKP * KKP;

    // Workspace layout
    char* ws = (char*)d_ws;
    __hip_bfloat16* w1b = (__hip_bfloat16*)(ws);                       // 786432 B
    __hip_bfloat16* w2b = (__hip_bfloat16*)(ws + 786432);              // 524288 B
    int* idx_ws = (int*)(ws + 1310720);                                // 3264 B
    int* sl_ws  = (int*)(ws + 1313984);                                // 32 B
    int* flags  = (int*)(ws + 1314016);                                // 245760 B

    // 1) weight conversion
    mp_prep_weights<<<(DD * HH + 255) / 256, 256, 0, stream>>>(w1, w2, w1b, w2b);

    // 2) fused MLP scorer (WMMA bf16): 960 blocks x 256 threads, 64 rows each
    mp_scorer<<<MTOT / MT, 256, 0, stream>>>(span_vecs, span_mask, w1b, b1, w2b, b2,
                                             w3, b3, o_scores);

    // 3) top-K rank flags: (B * 30) blocks
    mp_topk_rank<<<BB * 30, 256, 0, stream>>>(o_scores, flags);

    // 4) compaction + small outputs: B blocks
    mp_topk_compact<<<BB, 256, 0, stream>>>(flags, o_scores, seqlen, idx_ws, sl_ws,
                                            o_idx, o_fscore, o_fbegin, o_fend, o_slen);

    // 5) gather span vectors: B*K blocks
    mp_gather_vecs<<<BB * KKP, 256, 0, stream>>>(span_vecs, idx_ws, o_vecs);

    // 6) masks
    mp_masks<<<(BB * KKP * KKP + 255) / 256, 256, 0, stream>>>(sl_ws, o_sq, o_tri);
}
